// Decoder_6519760355498
// MI455X (gfx1250) — matrix operand order repaired and checked by reading, not compiled, not run on hardware
//
#include <hip/hip_runtime.h>
#include <hip/hip_bf16.h>

#define V_ 32000
#define E_ 512
#define H_ 512
#define B_ 32
#define T_ 128
#define L_ 32

typedef __bf16 v16bf __attribute__((ext_vector_type(16)));
typedef float  v8f   __attribute__((ext_vector_type(8)));

__device__ __forceinline__ unsigned short f2bf(float f) {
  unsigned u = __float_as_uint(f);
  u += 0x7FFFu + ((u >> 16) & 1u);          // round-to-nearest-even
  return (unsigned short)(u >> 16);
}

// ---------------- one-time kernels ----------------

__global__ void k_convert_bf16(const float* __restrict__ src,
                               unsigned short* __restrict__ dst, int n) {
  int i = blockIdx.x * blockDim.x + threadIdx.x;
  int stride = gridDim.x * blockDim.x;
  for (; i < n; i += stride) dst[i] = f2bf(src[i]);
}

// energies[b,t,d] = dot(W_attn[d,:], enc[b,t,:]) + b_attn[d]
__global__ void k_energies(const float* __restrict__ enc, const float* __restrict__ Wa,
                           const float* __restrict__ ba, float* __restrict__ eng) {
  __shared__ float row[H_];
  int bt = blockIdx.x;                       // 0..B*T-1
  const float* e = enc + (size_t)bt * H_;
  for (int i = threadIdx.x; i < H_; i += blockDim.x) row[i] = e[i];
  __syncthreads();
  for (int d = threadIdx.x; d < H_; d += blockDim.x) {
    const float* w = Wa + (size_t)d * H_;
    float acc = ba[d];
#pragma unroll 4
    for (int k = 0; k < H_; ++k) acc += w[k] * row[k];
    eng[(size_t)bt * H_ + d] = acc;
  }
}

__global__ void k_init(const int* __restrict__ inputs, const float* __restrict__ context,
                       const float* __restrict__ emb,
                       float* h, float* ctx, float* embd) {
  int i = blockIdx.x * blockDim.x + threadIdx.x;   // 0..B*H-1
  if (i >= B_ * H_) return;
  int b = i >> 9, d = i & 511;
  h[i] = 0.f;
  ctx[i] = context[i];
  embd[i] = emb[(size_t)inputs[b] * E_ + d];
}

// ---------------- per-step kernels ----------------

// gi[b,j] = b_ih[j] + W_ih[j,:].[embd|ctx] ; gh[b,j] = b_hh[j] + W_hh[j,:].h
__global__ void k_gates(const float* __restrict__ embd, const float* __restrict__ ctx,
                        const float* __restrict__ h,
                        const float* __restrict__ Wih, const float* __restrict__ Whh,
                        const float* __restrict__ bih, const float* __restrict__ bhh,
                        float* __restrict__ gi, float* __restrict__ gh) {
  __shared__ float x[1024];
  __shared__ float hr[512];
  int b = blockIdx.y;
  for (int k = threadIdx.x; k < 512; k += blockDim.x) {
    x[k]       = embd[b * 512 + k];
    x[512 + k] = ctx[b * 512 + k];
    hr[k]      = h[b * 512 + k];
  }
  __syncthreads();
  int j = blockIdx.x * blockDim.x + threadIdx.x;   // 0..1535
  if (j < 1536) {
    const float* wi = Wih + (size_t)j * 1024;
    float a = bih[j];
#pragma unroll 4
    for (int k = 0; k < 1024; ++k) a += wi[k] * x[k];
    gi[b * 1536 + j] = a;
    const float* wh = Whh + (size_t)j * 512;
    float c = bhh[j];
#pragma unroll 4
    for (int k = 0; k < 512; ++k) c += wh[k] * hr[k];
    gh[b * 1536 + j] = c;
  }
}

// GRU elementwise update + build bf16 A = [h_new | ctx_old] for the WMMA GEMM
__global__ void k_gru(const float* __restrict__ gi, const float* __restrict__ gh,
                      const float* __restrict__ ctx,
                      float* __restrict__ h, unsigned short* __restrict__ Abf) {
  int i = blockIdx.x * blockDim.x + threadIdx.x;
  if (i >= B_ * H_) return;
  int b = i >> 9, d = i & 511;
  const float* gib = gi + b * 1536;
  const float* ghb = gh + b * 1536;
  float r = 1.f / (1.f + expf(-(gib[d] + ghb[d])));
  float z = 1.f / (1.f + expf(-(gib[512 + d] + ghb[512 + d])));
  float n = tanhf(gib[1024 + d] + r * ghb[1024 + d]);
  float hn = (1.f - z) * n + z * h[i];
  h[i] = hn;
  Abf[b * 1024 + d]       = f2bf(hn);
  Abf[b * 1024 + 512 + d] = f2bf(ctx[i]);
}

// score[b, v] = A[b,:] . W_out[v,:]  via v_wmma_f32_16x16x32_bf16
// one wave = 32 rows x 16 vocab cols; 8 waves/block = 128 cols; grid 250 blocks
__global__ void __launch_bounds__(256)
k_score(const unsigned short* __restrict__ Abf, const unsigned short* __restrict__ Wb,
        const float* __restrict__ bias, float* __restrict__ out, int step) {
  __shared__ unsigned short sA[B_ * 1024];         // 64 KB
  const int tid = threadIdx.x;

  // async-stage A (32x1024 bf16 = 65536 B) into LDS: 16 x b128 per lane
  {
    unsigned sbase = (unsigned)(size_t)&sA[0];
    const char* g = (const char*)Abf;
#pragma unroll
    for (int i = 0; i < 16; ++i) {
      int c = tid + (i << 8);
      unsigned dst = sbase + c * 16;
      unsigned long long src = (unsigned long long)(g + c * 16);
      asm volatile("global_load_async_to_lds_b128 %0, %1, off"
                   :: "v"(dst), "v"(src) : "memory");
    }
    asm volatile("s_wait_asynccnt 0" ::: "memory");
  }
  __syncthreads();

  const int lane  = tid & 31;
  const int wave  = tid >> 5;
  const int vbase = blockIdx.x * 128 + wave * 16;
  const int hi    = (lane >= 16);
  const int v     = vbase + (lane & 15);
  const int koffA = hi ? 8 : 0;                    // ISA 16-bit A layout
  const int koffB = hi ? 8 : 0;                    // ISA 16-bit B layout
  const int rowA0 = (lane & 15);
  const int rowA1 = 16 + (lane & 15);

  union FragBF { uint4 q[2]; v16bf v; };
  v8f c0 = {}; v8f c1 = {};
  const uint4* wrow = (const uint4*)(Wb + (size_t)v * 1024);

  for (int k0 = 0; k0 < 1024; k0 += 32) {
    FragBF a0, a1, bb;
    a0.q[0] = *(const uint4*)&sA[rowA0 * 1024 + k0 + koffA];
    a0.q[1] = *(const uint4*)&sA[rowA0 * 1024 + k0 + 16 + koffA];
    a1.q[0] = *(const uint4*)&sA[rowA1 * 1024 + k0 + koffA];
    a1.q[1] = *(const uint4*)&sA[rowA1 * 1024 + k0 + 16 + koffA];
    int kb = k0 + koffB;
    bb.q[0] = wrow[(kb >> 3)];
    bb.q[1] = wrow[(kb >> 3) + 2];
    c0 = __builtin_amdgcn_wmma_f32_16x16x32_bf16(false, a0.v, false, bb.v,
                                                 (short)0, c0, false, false);
    c1 = __builtin_amdgcn_wmma_f32_16x16x32_bf16(false, a1.v, false, bb.v,
                                                 (short)0, c1, false, false);
  }

  float bv = bias[v];
  const int mbase = hi ? 8 : 0;                    // ISA f32 C/D layout
#pragma unroll
  for (int r = 0; r < 8; ++r) {
    int m0 = mbase + r;
    out[(size_t)m0 * (L_ * V_) + (size_t)step * V_ + v] = c0[r] + bv;
    int m1 = 16 + m0;
    out[(size_t)m1 * (L_ * V_) + (size_t)step * V_ + v] = c1[r] + bv;
  }
}

// fused per-row: streaming logsumexp + argmax + embedding gather for next step
__global__ void k_rowstats(const float* __restrict__ out, const float* __restrict__ emb,
                           float* __restrict__ stats, int* __restrict__ tok,
                           float* __restrict__ embd, int step) {
  __shared__ float sm[256], ss[256], am[256];
  __shared__ int   ai[256];
  __shared__ int   stok;
  int b = blockIdx.x, t = threadIdx.x;
  const float* row = out + (size_t)b * (L_ * V_) + (size_t)step * V_;
  float m = -3.4e38f, s = 0.f, bm = -3.4e38f; int bi = 0;
  for (int v2 = t; v2 < V_; v2 += 256) {
    float x = row[v2];
    if (x > m) { s = s * expf(m - x) + 1.f; m = x; }
    else       { s += expf(x - m); }
    if (x > bm) { bm = x; bi = v2; }
  }
  sm[t] = m; ss[t] = s; am[t] = bm; ai[t] = bi;
  __syncthreads();
  for (int off = 128; off > 0; off >>= 1) {
    if (t < off) {
      float m2 = sm[t + off], s2 = ss[t + off];
      if (m2 > sm[t]) { ss[t] = ss[t] * expf(sm[t] - m2) + s2; sm[t] = m2; }
      else            { ss[t] += s2 * expf(m2 - sm[t]); }
      float bm2 = am[t + off]; int bi2 = ai[t + off];
      if (bm2 > am[t] || (bm2 == am[t] && bi2 < ai[t])) { am[t] = bm2; ai[t] = bi2; }
    }
    __syncthreads();
  }
  if (t == 0) {
    stats[b * 2]     = sm[0];
    stats[b * 2 + 1] = ss[0];
    tok[b] = ai[0];
    stok   = ai[0];
  }
  __syncthreads();
  int tk = stok;
  for (int d = t; d < E_; d += 256) embd[b * E_ + d] = emb[(size_t)tk * E_ + d];
}

__global__ void k_logsm(float* __restrict__ out, const float* __restrict__ stats, int step) {
  int i = blockIdx.x * blockDim.x + threadIdx.x;   // 0..B*V-1
  if (i >= B_ * V_) return;
  int b = i / V_, v = i - b * V_;
  float lse = stats[b * 2] + logf(stats[b * 2 + 1]);
  size_t idx = (size_t)b * (L_ * V_) + (size_t)step * V_ + v;
  out[idx] = out[idx] - lse;
}

// attention: ae[t] = energies[b,t,:].h ; alpha = softmax(ae) ; ctx = alpha . enc
__global__ void k_attn(const float* __restrict__ eng, const float* __restrict__ enc,
                       const float* __restrict__ h, float* __restrict__ ctx) {
  __shared__ float hs[512];
  __shared__ float ae[128];
  __shared__ float red[128];
  int b = blockIdx.x, t = threadIdx.x;             // 128 threads
  for (int i = t; i < 512; i += 128) hs[i] = h[b * 512 + i];
  __syncthreads();
  const float* er = eng + ((size_t)b * T_ + t) * H_;
  float a = 0.f;
#pragma unroll 4
  for (int d = 0; d < 512; ++d) a += er[d] * hs[d];
  red[t] = a; __syncthreads();
  for (int off = 64; off > 0; off >>= 1) {
    if (t < off) red[t] = fmaxf(red[t], red[t + off]);
    __syncthreads();
  }
  float m = red[0]; __syncthreads();
  float ex = expf(a - m);
  red[t] = ex; __syncthreads();
  for (int off = 64; off > 0; off >>= 1) {
    if (t < off) red[t] += red[t + off];
    __syncthreads();
  }
  float s = red[0]; __syncthreads();
  ae[t] = ex / s;
  __syncthreads();
  for (int j = t; j < 512; j += 128) {
    float acc = 0.f;
#pragma unroll 4
    for (int tt = 0; tt < T_; ++tt) acc += ae[tt] * enc[((size_t)b * T_ + tt) * H_ + j];
    ctx[b * 512 + j] = acc;
  }
}

// ---------------- host launcher ----------------

extern "C" void kernel_launch(void* const* d_in, const int* in_sizes, int n_in,
                              void* d_out, int out_size, void* d_ws, size_t ws_size,
                              hipStream_t stream) {
  const int*   inputs  = (const int*)d_in[0];
  const float* context = (const float*)d_in[1];
  const float* enc     = (const float*)d_in[3];
  const float* emb     = (const float*)d_in[4];
  const float* Wih     = (const float*)d_in[5];
  const float* Whh     = (const float*)d_in[6];
  const float* bih     = (const float*)d_in[7];
  const float* bhh     = (const float*)d_in[8];
  const float* Wout    = (const float*)d_in[9];
  const float* bout    = (const float*)d_in[10];
  const float* Wattn   = (const float*)d_in[11];
  const float* battn   = (const float*)d_in[12];
  float* out = (float*)d_out;

  char* w = (char*)d_ws;
  auto alloc = [&](size_t bytes) {
    char* p = w;
    w += (bytes + 255) & ~(size_t)255;
    return p;
  };
  unsigned short* Wb  = (unsigned short*)alloc((size_t)V_ * 1024 * 2);  // 65.5 MB, L2-resident
  float* energies     = (float*)alloc((size_t)B_ * T_ * H_ * 4);        // 8.4 MB
  float* h            = (float*)alloc(B_ * H_ * 4);
  float* ctx          = (float*)alloc(B_ * H_ * 4);
  float* embd         = (float*)alloc(B_ * H_ * 4);
  float* gi           = (float*)alloc(B_ * 1536 * 4);
  float* gh           = (float*)alloc(B_ * 1536 * 4);
  unsigned short* Abf = (unsigned short*)alloc(B_ * 1024 * 2);
  float* stats        = (float*)alloc(B_ * 2 * 4);
  int*   tok          = (int*)alloc(B_ * 4);

  // one-time setup
  k_convert_bf16<<<8192, 256, 0, stream>>>(Wout, Wb, V_ * 1024);
  k_energies<<<B_ * T_, 256, 0, stream>>>(enc, Wattn, battn, energies);
  k_init<<<(B_ * H_ + 255) / 256, 256, 0, stream>>>(inputs, context, emb, h, ctx, embd);

  for (int s = 0; s < L_; ++s) {
    k_gates<<<dim3(6, B_), 256, 0, stream>>>(embd, ctx, h, Wih, Whh, bih, bhh, gi, gh);
    k_gru<<<(B_ * H_ + 255) / 256, 256, 0, stream>>>(gi, gh, ctx, h, Abf);
    k_score<<<V_ / 128, 256, 0, stream>>>(Abf, Wb, bout, out, s);
    k_rowstats<<<B_, 256, 0, stream>>>(out, emb, stats, tok, embd, s);
    k_logsm<<<(B_ * V_ + 255) / 256, 256, 0, stream>>>(out, stats, s);
    k_attn<<<B_, 128, 0, stream>>>(energies, enc, h, ctx);   // ctx for next step
  }
}
